// PSAttention_26663156973979
// MI455X (gfx1250) — matrix operand order repaired and checked by reading, not compiled, not run on hardware
//
#include <hip/hip_runtime.h>
#include <hip/hip_fp16.h>

typedef __attribute__((ext_vector_type(16))) _Float16 v16h;
typedef __attribute__((ext_vector_type(8)))  _Float16 v8h;
typedef __attribute__((ext_vector_type(8)))  float    v8f;
typedef __attribute__((ext_vector_type(4)))  unsigned int u32x4;
typedef __attribute__((ext_vector_type(8)))  unsigned int u32x8;

#define HH 128
#define WW 128
#define CC 32
#define FF 288           // 32 channels * 9 patch positions = 9 * 32 (K of wmma)
#define NN 16384         // HH*WW
#define KNN 8
#define NCHUNK 4         // column-space split for occupancy
#define TILES_PER_CHUNK (NN / 16 / NCHUNK)   // 256
#define USE_TDM 1        // stage B tiles with tensor_load_to_lds (TDM)

// ---------------------------------------------------------------------------
// Kernel 1: build f16 patch-feature matrices A (from q) and B (from k),
// plus f32 squared norms of the f16-rounded rows.
// feat[p][d*32+c] = img[c][clamp(y+dy-1)][clamp(x+dx-1)]   (edge padding)
// ---------------------------------------------------------------------------
__global__ void build_features(const float* __restrict__ q,
                               const float* __restrict__ k,
                               _Float16* __restrict__ Af,
                               _Float16* __restrict__ Bf,
                               float* __restrict__ na,
                               float* __restrict__ nb) {
  int p = blockIdx.x * blockDim.x + threadIdx.x;
  if (p >= NN) return;
  int y = p >> 7, x = p & (WW - 1);
  float sa = 0.f, sb = 0.f;
#pragma unroll
  for (int d = 0; d < 9; ++d) {
    int yy = y + d / 3 - 1;
    int xx = x + d % 3 - 1;
    yy = yy < 0 ? 0 : (yy > HH - 1 ? HH - 1 : yy);
    xx = xx < 0 ? 0 : (xx > WW - 1 ? WW - 1 : xx);
    int base = yy * WW + xx;
    for (int c = 0; c < CC; ++c) {
      _Float16 ah = (_Float16)q[c * NN + base];
      _Float16 bh = (_Float16)k[c * NN + base];
      Af[(size_t)p * FF + d * CC + c] = ah;
      Bf[(size_t)p * FF + d * CC + c] = bh;
      sa += (float)ah * (float)ah;
      sb += (float)bh * (float)bh;
    }
  }
  na[p] = sa;
  nb[p] = sb;
}

// ---------------------------------------------------------------------------
// Fragment loaders; A and B share one contiguous k order (k = k0+16*halfsel+s).
// A (16x32, MxK): lane l: M = l&15; halfsel = l>>4.
//   v16h elems 0..7  = K  k0+16*halfsel  .. +7      (contiguous 16B)
//   v16h elems 8..15 = K  k0+16*halfsel+8.. +7      (contiguous 16B)
// B (32x16, KxN): lane l: N = l&15; halfsel = l>>4.
//   v16h elems 0..15 = K  k0+16*halfsel .. +15      (contiguous 32B)
// ---------------------------------------------------------------------------
__device__ inline v16h load_fragA(const _Float16* __restrict__ Af, int row,
                                  int k0, int halfsel) {
  const v8h* p0 = (const v8h*)(Af + (size_t)row * FF + k0 + halfsel * 16);
  const v8h* p1 = (const v8h*)(Af + (size_t)row * FF + k0 + 8 + halfsel * 16);
  v8h lo = *p0, hi = *p1;
  return __builtin_shufflevector(lo, hi, 0, 1, 2, 3, 4, 5, 6, 7,
                                 8, 9, 10, 11, 12, 13, 14, 15);
}

__device__ inline v16h load_fragB_lds(const _Float16* ldsB, int n,
                                      int k0, int halfsel) {
  return *(const v16h*)(ldsB + (size_t)n * FF + k0 + halfsel * 16);
}

// ---------------------------------------------------------------------------
// Kernel 2: tiled GEMM cost matrix + streaming top-8 per row.
// Block = 128 threads = 4 waves; each wave owns a 16-row group.
// Grid.x = (NN/16/4) * NCHUNK = 1024.  Each block scans one column chunk.
// B tiles (16 cols x 288 f16 = 9216 B contiguous) are DMA'd to LDS by the
// Tensor Data Mover: wave 0 issues tensor_load_to_lds with a 1-D D#
// (data_size = 8 B, 1152 elements), waits on TENSORcnt, then all waves sync.
// ---------------------------------------------------------------------------
__global__ void __launch_bounds__(128)
gemm_topk(const _Float16* __restrict__ Af, const _Float16* __restrict__ Bf,
          const float* __restrict__ na, const float* __restrict__ nb,
          float* __restrict__ ptopc, int* __restrict__ ptopi) {
  __shared__ __align__(32) _Float16 ldsB[16 * FF];   // 9 KB staged B tile
  __shared__ float tile[4][256];                     // per-wave 16x16 costs

  const int lane = threadIdx.x & 31;
  const int wave = threadIdx.x >> 5;
  const int rowGroup = blockIdx.x >> 2;      // 0..255
  const int chunk = blockIdx.x & 3;          // 0..3
  const int rowBase = (rowGroup * 4 + wave) * 16;
  const int halfsel = lane >> 4;
  const int n = lane & 15;

  // Loop-invariant A fragments (9 K-chunks of 32) held in registers.
  v16h afrag[9];
#pragma unroll
  for (int kc = 0; kc < 9; ++kc)
    afrag[kc] = load_fragA(Af, rowBase + n, kc * 32, halfsel);

  float nam[8];
#pragma unroll
  for (int j = 0; j < 8; ++j) nam[j] = na[rowBase + j + halfsel * 8];

  float ck[KNN];
  int ik[KNN];
#pragma unroll
  for (int j = 0; j < KNN; ++j) { ck[j] = 3.0e38f; ik[j] = 0; }

#if USE_TDM
  // Loop-invariant D# pieces. Tile = 1152 x 8-byte elements (9216 B).
  const unsigned int lds_addr = (unsigned int)(uintptr_t)&ldsB[0];
  u32x8 g1;
  g1[0] = (3u << 16);          // workgroup_mask=0, data_size=3 (8B), no flags
  g1[1] = (1152u << 16);       // tensor_dim0[15:0] at group bits 63:48
  g1[2] = (1u << 16);          // tensor_dim0[31:16]=0 | tensor_dim1=1
  g1[3] = (1152u << 16);       // tensor_dim1[31:16]=0 | tile_dim0=1152
  g1[4] = 0u;                  // tile_dim1=0 (1-D), tile_dim2=0
  g1[5] = 1152u;               // tensor_dim0_stride[31:0]
  g1[6] = 0u;                  // stride[47:32]=0 | tensor_dim1_stride[15:0]=0
  g1[7] = 0u;                  // tensor_dim1_stride[47:16]=0
#endif

  for (int t = 0; t < TILES_PER_CHUNK; ++t) {
    const int colBase = (chunk * TILES_PER_CHUNK + t) * 16;

#if USE_TDM
    if (wave == 0) {
      const unsigned long long gaddr =
          (unsigned long long)(uintptr_t)(Bf + (size_t)colBase * FF);
      u32x4 g0;
      g0.x = 1u;                               // count=1, user mode, no gather
      g0.y = lds_addr;                         // D#.lds_addr
      g0.z = (unsigned int)gaddr;              // global_addr[31:0]
      g0.w = ((unsigned int)(gaddr >> 32) & 0x01FFFFFFu) | (2u << 30); // type=2
      asm volatile("tensor_load_to_lds %0, %1" :: "s"(g0), "s"(g1) : "memory");
      __builtin_amdgcn_s_wait_tensorcnt(0);
    }
    // Warm L2 for the next tile while this one is consumed.
    if (t + 1 < TILES_PER_CHUNK)
      __builtin_prefetch(
          (const void*)(Bf + (size_t)(colBase + 16) * FF + threadIdx.x * 32),
          0, 1);
#else
    {
      const uint4* src = (const uint4*)(Bf + (size_t)colBase * FF);
      uint4* dst = (uint4*)ldsB;
      for (int i = threadIdx.x; i < 16 * FF / 8; i += 128) {
        dst[i] = src[i];
        if (t + 1 < TILES_PER_CHUNK)
          __builtin_prefetch((const void*)(src + 16 * FF / 8 + i), 0, 1);
      }
    }
#endif
    __syncthreads();

    // 16x16x288 dot products: 9 accumulating WMMAs.
    v8f acc = {};
#pragma unroll
    for (int kc = 0; kc < 9; ++kc) {
      v16h bfrag = load_fragB_lds(ldsB, n, kc * 32, halfsel);
      acc = __builtin_amdgcn_wmma_f32_16x16x32_f16(
          false, afrag[kc], false, bfrag, (short)0, acc, false, false);
    }

    // cost(m,n) = ||A_m||^2 + ||B_n||^2 - 2 * dot  -> per-wave LDS tile.
    const float nbn = nb[colBase + n];
#pragma unroll
    for (int j = 0; j < 8; ++j)
      tile[wave][(j + halfsel * 8) * 16 + n] = nam[j] + nbn - 2.0f * acc[j];
    __syncthreads();

    // Lanes 0..15 each scan their row, maintaining a sorted top-8.
    if (lane < 16) {
#pragma unroll 1
      for (int cc = 0; cc < 16; ++cc) {
        float c = tile[wave][lane * 16 + cc];
        if (c < ck[KNN - 1]) {
          float cv = c;
          int iv = colBase + cc;
#pragma unroll
          for (int j = 0; j < KNN; ++j) {
            float tc = ck[j];
            int ti = ik[j];
            bool sw = cv < tc;
            ck[j] = sw ? cv : tc;
            ik[j] = sw ? iv : ti;
            cv = sw ? tc : cv;
            iv = sw ? ti : iv;
          }
        }
      }
    }
    __syncthreads();
  }

  if (lane < 16) {
    int row = rowBase + lane;
#pragma unroll
    for (int j = 0; j < KNN; ++j) {
      ptopc[((size_t)chunk * NN + row) * KNN + j] = ck[j];
      ptopi[((size_t)chunk * NN + row) * KNN + j] = ik[j];
    }
  }
}

// ---------------------------------------------------------------------------
// Kernel 3: merge 4 partial sorted top-8 lists per row into final top-8.
// ---------------------------------------------------------------------------
__global__ void merge_topk(const float* __restrict__ pc,
                           const int* __restrict__ pi,
                           float* __restrict__ topc, int* __restrict__ topi) {
  int p = blockIdx.x * blockDim.x + threadIdx.x;
  if (p >= NN) return;
  float ck[KNN];
  int ik[KNN];
#pragma unroll
  for (int j = 0; j < KNN; ++j) { ck[j] = 3.0e38f; ik[j] = 0; }
#pragma unroll 1
  for (int ch = 0; ch < NCHUNK; ++ch) {
#pragma unroll
    for (int j = 0; j < KNN; ++j) {
      float cv = pc[((size_t)ch * NN + p) * KNN + j];
      int iv = pi[((size_t)ch * NN + p) * KNN + j];
      if (cv < ck[KNN - 1]) {
#pragma unroll
        for (int q = 0; q < KNN; ++q) {
          float tc = ck[q];
          int ti = ik[q];
          bool sw = cv < tc;
          ck[q] = sw ? cv : tc;
          ik[q] = sw ? iv : ti;
          cv = sw ? tc : cv;
          iv = sw ? ti : iv;
        }
      }
    }
  }
#pragma unroll
  for (int j = 0; j < KNN; ++j) {
    topc[(size_t)p * KNN + j] = ck[j];
    topi[(size_t)p * KNN + j] = ik[j];
  }
}

// ---------------------------------------------------------------------------
// Kernel 4: per-pixel softmax over -cost (8 neighbors).
// ---------------------------------------------------------------------------
__global__ void softmax_w(const float* __restrict__ topc,
                          float* __restrict__ wgt) {
  int p = blockIdx.x * blockDim.x + threadIdx.x;
  if (p >= NN) return;
  float c[KNN], m = 3.0e38f;
#pragma unroll
  for (int j = 0; j < KNN; ++j) {
    c[j] = topc[(size_t)p * KNN + j];
    m = fminf(m, c[j]);
  }
  float e[KNN], s = 0.f;
#pragma unroll
  for (int j = 0; j < KNN; ++j) {
    e[j] = __expf(m - c[j]);   // softmax(-cost) with max-shift
    s += e[j];
  }
  float inv = 1.0f / s;
#pragma unroll
  for (int j = 0; j < KNN; ++j) wgt[(size_t)p * KNN + j] = e[j] * inv;
}

// ---------------------------------------------------------------------------
// Kernel 5: weighted gather of v at the top-8 shift indices.
// out[(b*64+c)][p] = sum_j w[p][j] * v[(b*64+c)][idx[p][j]]
// ---------------------------------------------------------------------------
__global__ void gather_out(const float* __restrict__ v,
                           const float* __restrict__ wgt,
                           const int* __restrict__ topi,
                           float* __restrict__ out) {
  int t = blockIdx.x * blockDim.x + threadIdx.x;
  if (t >= 4 * 64 * NN) return;
  int p = t & (NN - 1);
  int bc = t >> 14;
  const float* vp = v + (size_t)bc * NN;
  float s = 0.f;
#pragma unroll
  for (int j = 0; j < KNN; ++j)
    s += wgt[(size_t)p * KNN + j] * vp[topi[(size_t)p * KNN + j]];
  out[t] = s;
}

// ---------------------------------------------------------------------------
extern "C" void kernel_launch(void* const* d_in, const int* in_sizes, int n_in,
                              void* d_out, int out_size, void* d_ws,
                              size_t ws_size, hipStream_t stream) {
  const float* q = (const float*)d_in[0];   // (1,32,128,128)
  const float* k = (const float*)d_in[1];   // (1,32,128,128)
  const float* v = (const float*)d_in[2];   // (4,64,128,128)
  float* out = (float*)d_out;               // (4,64,128,128)

  char* ws = (char*)d_ws;
  size_t off = 0;
  auto carve = [&](size_t bytes) -> void* {
    void* p = (void*)(ws + off);
    off = (off + bytes + 255) & ~(size_t)255;
    return p;
  };
  _Float16* Af = (_Float16*)carve((size_t)NN * FF * 2);
  _Float16* Bf = (_Float16*)carve((size_t)NN * FF * 2);
  float* na = (float*)carve((size_t)NN * 4);
  float* nb = (float*)carve((size_t)NN * 4);
  float* ptopc = (float*)carve((size_t)NCHUNK * NN * KNN * 4);
  int* ptopi = (int*)carve((size_t)NCHUNK * NN * KNN * 4);
  float* topc = (float*)carve((size_t)NN * KNN * 4);
  int* topi = (int*)carve((size_t)NN * KNN * 4);
  float* wgt = (float*)carve((size_t)NN * KNN * 4);
  (void)ws_size; (void)in_sizes; (void)n_in; (void)out_size;

  build_features<<<NN / 256, 256, 0, stream>>>(q, k, Af, Bf, na, nb);
  gemm_topk<<<(NN / 16 / 4) * NCHUNK, 128, 0, stream>>>(Af, Bf, na, nb,
                                                        ptopc, ptopi);
  merge_topk<<<NN / 256, 256, 0, stream>>>(ptopc, ptopi, topc, topi);
  softmax_w<<<NN / 256, 256, 0, stream>>>(topc, wgt);
  gather_out<<<(4 * 64 * NN) / 256, 256, 0, stream>>>(v, wgt, topi, out);
}
